// PCNet_34978213658816
// MI455X (gfx1250) — matrix operand order repaired and checked by reading, not compiled, not run on hardware
//
#include <hip/hip_runtime.h>
#include <hip/hip_bf16.h>
#include <float.h>
#include <stdint.h>

#define B_GRAPHS 8192
#define N_USERS  128
#define N_SERVERS 8
#define OUT_DIM  1024                 // N_USERS * N_SERVERS
#define IN_REAL  1160
#define IN_PAD   1184                 // padded to multiple of 32
#define HID      256
#define OUT4     3072                 // 3 * OUT_DIM
#define NSEG     (B_GRAPHS * N_USERS) // 1,048,576
#define NEDGE    (B_GRAPHS * OUT_DIM) // 8,388,608

typedef __attribute__((ext_vector_type(16))) __bf16 v16bf;
typedef __attribute__((ext_vector_type(8)))  __bf16 v8bf;
typedef __attribute__((ext_vector_type(8)))  float  v8f;

// GCC-style int4 to exactly match the async-LDS builtin's parameter type
typedef int v4i_ __attribute__((vector_size(16)));
typedef __attribute__((address_space(1))) v4i_* gv4p;
typedef __attribute__((address_space(3))) v4i_* lv4p;

#if defined(__HIP_DEVICE_COMPILE__) && __has_builtin(__builtin_amdgcn_global_load_async_to_lds_b128)
#define HAS_ASYNC_LDS 1
#else
#define HAS_ASYNC_LDS 0
#endif

// ---------- helpers ----------
__device__ __forceinline__ unsigned fenc(float f) {
    unsigned u = __float_as_uint(f);
    return (u & 0x80000000u) ? ~u : (u | 0x80000000u);   // monotonic order-preserving map
}
__device__ __forceinline__ float fdec(unsigned u) {
    u = (u & 0x80000000u) ? (u & 0x7FFFFFFFu) : ~u;
    return __uint_as_float(u);
}
__device__ __forceinline__ v16bf combine16(v8bf lo, v8bf hi) {
    return __builtin_shufflevector(lo, hi, 0,1,2,3,4,5,6,7,8,9,10,11,12,13,14,15);
}
// A fragment (16x32 bf16): lane half 'hi': VGPRs0-3 = K hi*8..hi*8+7, VGPRs4-7 = K 16+hi*8..
__device__ __forceinline__ v16bf load_a_frag(const __bf16* rowp, int hi) {
    v8bf lo = *(const v8bf*)(rowp + hi * 8);
    v8bf h2 = *(const v8bf*)(rowp + 16 + hi * 8);
    return combine16(lo, h2);
}
// B fragment (32x16 bf16, staged as W^T rows): lane half 'hi': VGPRs0-3 = K hi*8..hi*8+7, VGPRs4-7 = K 16+hi*8..
__device__ __forceinline__ v16bf load_b_frag(const __bf16* rowp, int hi) {
    v8bf lo = *(const v8bf*)(rowp + hi * 8);
    v8bf h2 = *(const v8bf*)(rowp + 16 + hi * 8);
    return combine16(lo, h2);
}

// 16-byte global -> LDS copy: async (ASYNCcnt / global_load_async_to_lds_b128) when
// available, else through-VGPR fallback.
// Flat->AS pointers formed via uintptr_t: global flat VA == global VA; for LDS the
// flat address's low 32 bits are the LDS offset (ISA 10.2 aperture rules).
__device__ __forceinline__ void cp16_g2l(const __bf16* g, __bf16* l) {
#if HAS_ASYNC_LDS
    __builtin_amdgcn_global_load_async_to_lds_b128(
        (gv4p)(uintptr_t)g, (lv4p)(uintptr_t)l, 0, 0);
#else
    *(v8bf*)l = *(const v8bf*)g;
#endif
}
__device__ __forceinline__ void wait_async_tile_inflight() {
#if HAS_ASYNC_LDS
#if __has_builtin(__builtin_amdgcn_s_wait_asynccnt)
    __builtin_amdgcn_s_wait_asynccnt(3);
#else
    asm volatile("s_wait_asynccnt 0x3" ::: "memory");
#endif
#endif
}
__device__ __forceinline__ void wait_async_all() {
#if HAS_ASYNC_LDS
#if __has_builtin(__builtin_amdgcn_s_wait_asynccnt)
    __builtin_amdgcn_s_wait_asynccnt(0);
#else
    asm volatile("s_wait_asynccnt 0x0" ::: "memory");
#endif
#endif
}

// ---------- feature packing: [B, IN_PAD] bf16, zero padded ----------
__global__ void pack_feats_kernel(const float* __restrict__ pl,
                                  const float* __restrict__ ut,
                                  const float* __restrict__ sc,
                                  __bf16* __restrict__ feats) {
    size_t idx = (size_t)blockIdx.x * blockDim.x + threadIdx.x;
    if (idx >= (size_t)B_GRAPHS * IN_PAD) return;
    int row = (int)(idx / IN_PAD);
    int col = (int)(idx % IN_PAD);
    float v = 0.f;
    if (col < OUT_DIM)                   v = pl[(size_t)row * OUT_DIM + col];
    else if (col < OUT_DIM + N_USERS)    v = ut[(size_t)row * N_USERS + (col - OUT_DIM)];
    else if (col < IN_REAL)              v = sc[(size_t)row * N_SERVERS + (col - OUT_DIM - N_USERS)];
    feats[idx] = (__bf16)v;
}

// ---------- weight transpose+convert: W[Kreal x N] f32 -> WT[N x Kpad] bf16 (zero pad) ----------
__global__ void conv_wT_kernel(const float* __restrict__ W, __bf16* __restrict__ WT,
                               int N, int Kpad, int Kreal) {
    size_t idx = (size_t)blockIdx.x * blockDim.x + threadIdx.x;
    if (idx >= (size_t)N * Kpad) return;
    int n = (int)(idx / Kpad);
    int k = (int)(idx % Kpad);
    WT[idx] = (k < Kreal) ? (__bf16)W[(size_t)k * N + n] : (__bf16)0.f;
}

// ---------- WMMA GEMM: C = act(A[8192 x K] @ W[K x N] + bias) ----------
// A bf16 row-major (lda=K), WT bf16 [N][K]; block tile 128x64, wave tile 32x32 (2x2 frags).
// Double-buffered LDS; tiles streamed with async global->LDS (ASYNCcnt) when available.
template <bool FINAL>
__global__ __launch_bounds__(256)
void gemm_wmma_kernel(const __bf16* __restrict__ A, const __bf16* __restrict__ WT,
                      const float* __restrict__ bias, __bf16* __restrict__ outB,
                      float* __restrict__ outF, int K, int N) {
    __shared__ __align__(16) __bf16 lA[2][128 * 40];
    __shared__ __align__(16) __bf16 lB[2][64 * 40];

    const int tid  = threadIdx.x;
    const int lane = tid & 31;
    const int wid  = tid >> 5;           // 0..7
    const int rowBase = blockIdx.y * 128;
    const int colBase = blockIdx.x * 64;
    const int wRow = (wid >> 1) * 32;    // 0,32,64,96
    const int wCol = (wid & 1) * 32;     // 0,32
    const int lr = lane & 15;
    const int hi = lane >> 4;

    // per-thread staging assignment: 3 x 16B chunks per tile
    const int ra0 = tid >> 2;            // A rows 0..63
    const int ra1 = ra0 + 64;            // A rows 64..127
    const int kc  = (tid & 3) * 8;       // 0,8,16,24

    v8f acc[2][2] = {};

    auto stage = [&](int k0, int buf) {
        cp16_g2l(&A [(size_t)(rowBase + ra0) * K + k0 + kc], &lA[buf][ra0 * 40 + kc]);
        cp16_g2l(&A [(size_t)(rowBase + ra1) * K + k0 + kc], &lA[buf][ra1 * 40 + kc]);
        cp16_g2l(&WT[(size_t)(colBase + ra0) * K + k0 + kc], &lB[buf][ra0 * 40 + kc]);
    };

    const int nk = K >> 5;
    stage(0, 0);

    for (int i = 0; i < nk; ++i) {
        const int buf = i & 1;
        const bool more = (i + 1) < nk;
        if (more) {
            stage((i + 1) << 5, buf ^ 1);     // stream next tile into other buffer
            __builtin_prefetch(&A[(size_t)(rowBase + ra0) * K + ((i + 2) << 5)], 0, 1);
            wait_async_tile_inflight();       // current tile done; next tile (3 issues) in flight
        } else {
            wait_async_all();
        }
        __syncthreads();

        v16bf af[2], bfr[2];
        #pragma unroll
        for (int x = 0; x < 2; ++x)
            af[x] = load_a_frag(&lA[buf][(wRow + x * 16 + lr) * 40], hi);
        #pragma unroll
        for (int j = 0; j < 2; ++j)
            bfr[j] = load_b_frag(&lB[buf][(wCol + j * 16 + lr) * 40], hi);

        #pragma unroll
        for (int x = 0; x < 2; ++x)
            #pragma unroll
            for (int j = 0; j < 2; ++j)
                acc[x][j] = __builtin_amdgcn_wmma_f32_16x16x32_bf16(
                    false, af[x], false, bfr[j], (short)0, acc[x][j], false, false);
        __syncthreads();                      // all reads of this buffer done before it is re-filled
    }

    // write out: C layout -> M = hi*8+g, N = lr
    #pragma unroll
    for (int x = 0; x < 2; ++x) {
        #pragma unroll
        for (int j = 0; j < 2; ++j) {
            #pragma unroll
            for (int g = 0; g < 8; ++g) {
                int r  = rowBase + wRow + x * 16 + hi * 8 + g;
                int cN = colBase + wCol + j * 16 + lr;
                float v = acc[x][j][g] + bias[cN];
                if (!FINAL) {
                    v = v > 0.f ? v : 0.f;
                    outB[(size_t)r * N + cN] = (__bf16)v;
                } else {
                    int blk = cN >> 10, within = cN & 1023;
                    outF[(size_t)blk * B_GRAPHS * OUT_DIM + (size_t)r * OUT_DIM + within] = v;
                }
            }
        }
    }
}

// ---------- segment softmax passes ----------
__global__ void seg_max_kernel(const float* __restrict__ x, const long long* __restrict__ ei,
                               unsigned* __restrict__ segm) {
    size_t e = (size_t)blockIdx.x * blockDim.x + threadIdx.x;
    if (e >= NEDGE) return;
    long long i0 = ei[e];
    long long i1 = ei[(size_t)NEDGE + e];
    atomicMax(&segm[i0],                  fenc(x[e]));
    atomicMax(&segm[(size_t)NSEG + i0],   fenc(x[(size_t)NEDGE + e]));
    atomicMax(&segm[2*(size_t)NSEG + i1], fenc(x[2*(size_t)NEDGE + e]));
}

__global__ void seg_sum_kernel(const float* __restrict__ x, const long long* __restrict__ ei,
                               const unsigned* __restrict__ segm, float* __restrict__ segs) {
    size_t e = (size_t)blockIdx.x * blockDim.x + threadIdx.x;
    if (e >= NEDGE) return;
    long long i0 = ei[e];
    long long i1 = ei[(size_t)NEDGE + e];
    atomicAdd(&segs[i0],                  expf(x[e]                   - fdec(segm[i0])));
    atomicAdd(&segs[(size_t)NSEG + i0],   expf(x[(size_t)NEDGE + e]   - fdec(segm[(size_t)NSEG + i0])));
    atomicAdd(&segs[2*(size_t)NSEG + i1], expf(x[2*(size_t)NEDGE + e] - fdec(segm[2*(size_t)NSEG + i1])));
}

__global__ void seg_fin_kernel(float* __restrict__ x, const long long* __restrict__ ei,
                               const unsigned* __restrict__ segm, const float* __restrict__ segs) {
    size_t e = (size_t)blockIdx.x * blockDim.x + threadIdx.x;
    if (e >= NEDGE) return;
    long long i0 = ei[e];
    long long i1 = ei[(size_t)NEDGE + e];
    x[e] = expf(x[e] - fdec(segm[i0])) / segs[i0];
    x[(size_t)NEDGE + e] =
        expf(x[(size_t)NEDGE + e] - fdec(segm[(size_t)NSEG + i0])) / segs[(size_t)NSEG + i0];
    x[2*(size_t)NEDGE + e] =
        expf(x[2*(size_t)NEDGE + e] - fdec(segm[2*(size_t)NSEG + i1])) / segs[2*(size_t)NSEG + i1];
}

// ---------- launch ----------
extern "C" void kernel_launch(void* const* d_in, const int* in_sizes, int n_in,
                              void* d_out, int out_size, void* d_ws, size_t ws_size,
                              hipStream_t stream) {
    const float* pl = (const float*)d_in[0];
    const float* ut = (const float*)d_in[1];
    const float* sc = (const float*)d_in[2];
    const long long* ei = (const long long*)d_in[3];
    const float* W1 = (const float*)d_in[4];  const float* b1 = (const float*)d_in[5];
    const float* W2 = (const float*)d_in[6];  const float* b2 = (const float*)d_in[7];
    const float* W3 = (const float*)d_in[8];  const float* b3 = (const float*)d_in[9];
    const float* W4 = (const float*)d_in[10]; const float* b4 = (const float*)d_in[11];
    float* out = (float*)d_out;

    char* ws = (char*)d_ws;
    size_t off = 0;
    auto take = [&](size_t bytes) -> char* {
        char* p = ws + off;
        off += (bytes + 255) & ~(size_t)255;
        return p;
    };
    __bf16* feats = (__bf16*)take((size_t)B_GRAPHS * IN_PAD * 2);
    __bf16* W1T   = (__bf16*)take((size_t)HID  * IN_PAD * 2);
    __bf16* W2T   = (__bf16*)take((size_t)HID  * HID    * 2);
    __bf16* W3T   = (__bf16*)take((size_t)HID  * HID    * 2);
    __bf16* W4T   = (__bf16*)take((size_t)OUT4 * HID    * 2);
    __bf16* h1    = (__bf16*)take((size_t)B_GRAPHS * HID * 2);
    __bf16* h2    = (__bf16*)take((size_t)B_GRAPHS * HID * 2);
    __bf16* h3    = (__bf16*)take((size_t)B_GRAPHS * HID * 2);
    unsigned* segm = (unsigned*)take((size_t)3 * NSEG * 4);
    float*    segs = (float*)take((size_t)3 * NSEG * 4);

    // zero the segment-max (monotonic-uint: 0 < any finite value) and segment-sum arrays
    (void)hipMemsetAsync(segm, 0, (size_t)3 * NSEG * 4, stream);
    (void)hipMemsetAsync(segs, 0, (size_t)3 * NSEG * 4, stream);

    // pack + convert
    {
        size_t n = (size_t)B_GRAPHS * IN_PAD;
        pack_feats_kernel<<<(unsigned)((n + 255) / 256), 256, 0, stream>>>(pl, ut, sc, feats);
    }
    {
        size_t n = (size_t)HID * IN_PAD;
        conv_wT_kernel<<<(unsigned)((n + 255) / 256), 256, 0, stream>>>(W1, W1T, HID, IN_PAD, IN_REAL);
        n = (size_t)HID * HID;
        conv_wT_kernel<<<(unsigned)((n + 255) / 256), 256, 0, stream>>>(W2, W2T, HID, HID, HID);
        conv_wT_kernel<<<(unsigned)((n + 255) / 256), 256, 0, stream>>>(W3, W3T, HID, HID, HID);
        n = (size_t)OUT4 * HID;
        conv_wT_kernel<<<(unsigned)((n + 255) / 256), 256, 0, stream>>>(W4, W4T, OUT4, HID, HID);
    }

    // MLP: 4 WMMA GEMMs
    dim3 blk(256);
    gemm_wmma_kernel<false><<<dim3(HID / 64,  B_GRAPHS / 128), blk, 0, stream>>>(
        feats, W1T, b1, h1, nullptr, IN_PAD, HID);
    gemm_wmma_kernel<false><<<dim3(HID / 64,  B_GRAPHS / 128), blk, 0, stream>>>(
        h1, W2T, b2, h2, nullptr, HID, HID);
    gemm_wmma_kernel<false><<<dim3(HID / 64,  B_GRAPHS / 128), blk, 0, stream>>>(
        h2, W3T, b3, h3, nullptr, HID, HID);
    gemm_wmma_kernel<true><<<dim3(OUT4 / 64, B_GRAPHS / 128), blk, 0, stream>>>(
        h3, W4T, b4, nullptr, out, HID, OUT4);

    // segment softmax (3 passes), normalizing d_out in place
    unsigned eg = (NEDGE + 255) / 256;
    seg_max_kernel<<<eg, 256, 0, stream>>>(out, ei, segm);
    seg_sum_kernel<<<eg, 256, 0, stream>>>(out, ei, segm, segs);
    seg_fin_kernel<<<eg, 256, 0, stream>>>(out, ei, segm, segs);
}
